// TF1CompatModel_87703232185001
// MI455X (gfx1250) — matrix operand order repaired and checked by reading, not compiled, not run on hardware
//
#include <hip/hip_runtime.h>
#include <hip/hip_bf16.h>
#include <stddef.h>

// ---------------------------------------------------------------------------
// Graves handwriting RNN on gfx1250 (CDNA5, wave32, WMMA bf16 16x16x32).
//
// One workgroup (256 thr = 8 waves) owns a 16-row batch tile and runs all
// T=800 steps internally.  Weights are prepacked to bf16 in the exact WMMA
// B-fragment lane/slot layout; activations are kept in LDS in the WMMA
// A-fragment layout.  Biases are folded into the GEMM via a constant-1.0
// K column.  c-state lives in VGPRs (v8f per unit-tile).
// ---------------------------------------------------------------------------

typedef __attribute__((ext_vector_type(16))) __bf16 v16bf;
typedef __attribute__((ext_vector_type(8)))  float  v8f;

#define RNNU   400
#define KWIN   10
#define ALPHAN 80
#define BTOT   256
#define TLEN   800
#define ULEN   100
#define MDNOUT 121

// Padded K-vector layouts (32-elem k-tiles):
//  V1 (len 448, 14 kt): [x(0..2), 1.0@3, 0.., h1@32..431, 0..]
//  V2 (len 928, 29 kt): [h1@0..399, 1.0@400, 0.., win@416..495, x@496..498,
//                        1.0@499, 0.., h2@512..911, 1.0(for mdn slot unused), 0..]
//  V3 (len 928, 29 kt): [h2@0..399, 0.., win@416..495, x@496..498, 1.0@499,
//                        0.., h3@512..911, 1.0@912, 0..]
#define NKT1  14
#define NKT23 29

__device__ __forceinline__ float sigf(float x) { return 1.f / (1.f + __expf(-x)); }

__device__ __forceinline__ v8f wmma_bf16(v16bf a, v16bf b, v8f c) {
  return __builtin_amdgcn_wmma_f32_16x16x32_bf16(false, a, false, b, (short)0, c,
                                                 false, false);
}

// Write scalar (row m, global K index) into A-fragment-swizzled LDS buffer.
// A 16-bit 16x32 layout: lane = m + 16*khi, khi=(K&8)?1:0 within tile,
// slot = (K&7) | ((K&16)?8:0).
__device__ __forceinline__ void st_bf(__bf16* XA, int K, int m, float v) {
  int kt   = K >> 5;
  int kk   = K & 31;
  int khi  = (kk >> 3) & 1;
  int slot = (kk & 7) | ((kk & 16) ? 8 : 0);
  XA[(((kt * 32) + m + 16 * khi) << 4) + slot] = (__bf16)v;
}

// ---------------------------------------------------------------------------
// Prepack kernels: f32 weights -> bf16 B-fragments.
// PW layout: [ut(25)][gate(4)][kt][lane(32)][slot(16)],
// element (lane,slot) holds W[col = ut*16 + (lane&15)][K = kt*32 + (slot&7) + 8*(lane>>4) + 16*(slot>>3)]
// ---------------------------------------------------------------------------
__global__ void pack_lstm_k(void* pwv, const float* __restrict__ w_ih,
                            const float* __restrict__ w_hh,
                            const float* __restrict__ b, int nkt, int mode) {
  __bf16* pw = (__bf16*)pwv;
  int total = 25 * 4 * nkt * 512;
  for (int i = blockIdx.x * blockDim.x + threadIdx.x; i < total;
       i += gridDim.x * blockDim.x) {
    int slot = i & 15;
    int lane = (i >> 4) & 31;
    int rest = i >> 9;                // (ut*4+g)*nkt + kt
    int kt = rest % nkt;
    int gg = rest / nkt;
    int g = gg & 3;
    int ut = gg >> 2;
    int u = ut * 16 + (lane & 15);
    int r = g * 400 + u;              // row in the 1600-gate weight
    int K = kt * 32 + (slot & 7) + 8 * (lane >> 4) + ((slot & 8) ? 16 : 0);
    float v = 0.f;
    if (mode == 1) {                  // layer 1: K map for V1
      if (K < 3)                         v = w_ih[r * 3 + K];
      else if (K == 3)                   v = b[r];
      else if (K >= 32 && K < 432)       v = w_hh[r * 400 + (K - 32)];
    } else {                          // layers 2/3: K map for V2/V3
      if (K < 400)                       v = w_ih[r * 483 + K];
      else if (K >= 416 && K < 496)      v = w_ih[r * 483 + 400 + (K - 416)];
      else if (K >= 496 && K < 499)      v = w_ih[r * 483 + 480 + (K - 496)];
      else if (K == 499)                 v = b[r];
      else if (K >= 512 && K < 912)      v = w_hh[r * 400 + (K - 512)];
    }
    pw[i] = (__bf16)v;
  }
}

// w_win: 2 N-tiles x 13 k-tiles over V2 K=0..415; bias at K=400.
__global__ void pack_win_k(void* pwv, const float* __restrict__ w_win,
                           const float* __restrict__ b_win) {
  __bf16* pw = (__bf16*)pwv;
  int total = 2 * 13 * 512;
  for (int i = blockIdx.x * blockDim.x + threadIdx.x; i < total;
       i += gridDim.x * blockDim.x) {
    int slot = i & 15;
    int lane = (i >> 4) & 31;
    int rest = i >> 9;
    int ktl = rest % 13;
    int nt = rest / 13;
    int n = nt * 16 + (lane & 15);
    int K = ktl * 32 + (slot & 7) + 8 * (lane >> 4) + ((slot & 8) ? 16 : 0);
    float v = 0.f;
    if (n < 30) {
      if (K < 400)       v = w_win[n * 400 + K];
      else if (K == 400) v = b_win[n];
    }
    pw[i] = (__bf16)v;
  }
}

// w_mdn: 8 N-tiles x 13 k-tiles over V3 K'=0..415 (h3 region, K=512+K'); bias at K'=400.
__global__ void pack_mdn_k(void* pwv, const float* __restrict__ w_mdn,
                           const float* __restrict__ b_mdn) {
  __bf16* pw = (__bf16*)pwv;
  int total = 8 * 13 * 512;
  for (int i = blockIdx.x * blockDim.x + threadIdx.x; i < total;
       i += gridDim.x * blockDim.x) {
    int slot = i & 15;
    int lane = (i >> 4) & 31;
    int rest = i >> 9;
    int ktl = rest % 13;
    int nt = rest / 13;
    int n = nt * 16 + (lane & 15);
    int K = ktl * 32 + (slot & 7) + 8 * (lane >> 4) + ((slot & 8) ? 16 : 0);
    float v = 0.f;
    if (n < MDNOUT) {
      if (K < 400)       v = w_mdn[n * 400 + K];
      else if (K == 400) v = b_mdn[n];
    }
    pw[i] = (__bf16)v;
  }
}

// char_sequence [B][U][A] -> csT [B][A][U]  (contiguous in U for window matvec)
__global__ void tr_char_k(float* __restrict__ csT, const float* __restrict__ cs) {
  int total = BTOT * ULEN * ALPHAN;
  for (int i = blockIdx.x * blockDim.x + threadIdx.x; i < total;
       i += gridDim.x * blockDim.x) {
    int u = i % ULEN;
    int a = (i / ULEN) % ALPHAN;
    int b = i / (ULEN * ALPHAN);
    csT[i] = cs[((size_t)b * ULEN + u) * ALPHAN + a];
  }
}

// ---------------------------------------------------------------------------
// LSTM stage: gates[16,1600] = XA(K) x PW, nonlinearity, c in regs, h -> LDS.
// Deferred h write behind a barrier avoids racing readers of the recurrent K
// region of the same buffer.
// ---------------------------------------------------------------------------
template <int NKT>
__device__ __forceinline__ void lstm_stage(const __bf16* XAsrc, const __bf16* PW,
                                           __bf16* hdst0, int hbase0,
                                           __bf16* hdst1, int hbase1,
                                           v8f* cst, int lane, int wave) {
  v8f hv[4];
#pragma unroll
  for (int tt = 0; tt < 4; ++tt) {
    int ut = wave + tt * 8;
    if (ut < 25) {
      v8f a0 = {0.f, 0.f, 0.f, 0.f, 0.f, 0.f, 0.f, 0.f};
      v8f a1 = a0, a2 = a0, a3 = a0;
      for (int kt = 0; kt < NKT; ++kt) {
        v16bf afr = *(const v16bf*)(XAsrc + (((kt * 32) + lane) << 4));
        const __bf16* pb =
            PW + ((size_t)(((ut * 4) * NKT + kt) * 32 + lane) << 4);
        v16bf b0 = *(const v16bf*)(pb);
        v16bf b1 = *(const v16bf*)(pb + (size_t)NKT * 512);
        v16bf b2 = *(const v16bf*)(pb + (size_t)NKT * 1024);
        v16bf b3 = *(const v16bf*)(pb + (size_t)NKT * 1536);
        a0 = wmma_bf16(afr, b0, a0);
        a1 = wmma_bf16(afr, b1, a1);
        a2 = wmma_bf16(afr, b2, a2);
        a3 = wmma_bf16(afr, b3, a3);
      }
      v8f cold = cst[tt], cnew, h;
#pragma unroll
      for (int e = 0; e < 8; ++e) {
        float ig = sigf(a0[e]);
        float fg = sigf(a1[e]);
        float gg = tanhf(a2[e]);
        float og = sigf(a3[e]);
        float cv = fg * cold[e] + ig * gg;
        cnew[e] = cv;
        h[e] = og * tanhf(cv);
      }
      cst[tt] = cnew;
      hv[tt] = h;
    }
  }
  __syncthreads();
#pragma unroll
  for (int tt = 0; tt < 4; ++tt) {
    int ut = wave + tt * 8;
    if (ut < 25) {
      int hi = lane >> 4;
      int u = ut * 16 + (lane & 15);
#pragma unroll
      for (int e = 0; e < 8; ++e) {
        int m = e + 8 * hi;
        float v = hv[tt][e];
        st_bf(hdst0, hbase0 + u, m, v);
        if (hdst1) st_bf(hdst1, hbase1 + u, m, v);
      }
    }
  }
  __syncthreads();
}

// ---------------------------------------------------------------------------
// Main persistent kernel: 16 workgroups x 256 threads, 800 steps each.
// ---------------------------------------------------------------------------
__global__ __launch_bounds__(256, 1) void rnn_main(
    const float* __restrict__ traj, const float* __restrict__ csT,
    const void* pw1v, const void* pw2v, const void* pw3v, const void* pwwinv,
    const void* pwmdnv, float* __restrict__ out) {
  const __bf16* PW1 = (const __bf16*)pw1v;
  const __bf16* PW2 = (const __bf16*)pw2v;
  const __bf16* PW3 = (const __bf16*)pw3v;
  const __bf16* PWW = (const __bf16*)pwwinv;
  const __bf16* PWM = (const __bf16*)pwmdnv;

  extern __shared__ __align__(128) char smem[];
  __bf16* XA1 = (__bf16*)smem;             // 14*512 bf16
  __bf16* XA2 = XA1 + NKT1 * 512;          // 29*512
  __bf16* XA3 = XA2 + NKT23 * 512;         // 29*512
  float* sAlpha = (float*)(smem + (NKT1 + 2 * NKT23) * 512 * 2);  // 16*10
  float* sBeta = sAlpha + 160;
  float* sKappa = sBeta + 160;
  float* sPhi = sKappa + 160;              // 16*100

  const int tid = threadIdx.x;
  const int lane = tid & 31;
  const int wave = tid >> 5;
  const int bbase = blockIdx.x * 16;

  // init: zero all activation fragments, alpha/beta/kappa
  for (int i = tid; i < (NKT1 + 2 * NKT23) * 512; i += 256) XA1[i] = (__bf16)0.f;
  for (int i = tid; i < 480; i += 256) sAlpha[i] = 0.f;
  // constant-1.0 bias columns (persist: never overwritten)
  if (tid < 16) {
    int m = tid;
    st_bf(XA1, 3, m, 1.f);
    st_bf(XA2, 400, m, 1.f);   // b_win hook
    st_bf(XA2, 499, m, 1.f);   // b2 hook
    st_bf(XA3, 499, m, 1.f);   // b3 hook
    st_bf(XA3, 912, m, 1.f);   // b_mdn hook
  }
  const v8f vz = {0.f, 0.f, 0.f, 0.f, 0.f, 0.f, 0.f, 0.f};
  v8f c1[4], c2[4], c3[4];
#pragma unroll
  for (int q = 0; q < 4; ++q) { c1[q] = vz; c2[q] = vz; c3[q] = vz; }
  __syncthreads();

  for (int t = 0; t < TLEN; ++t) {
    // ---- stage x_t into all three fragment buffers -----------------------
    if (tid < 48) {
      int m = tid / 3, cc = tid - m * 3;
      float x = traj[(((size_t)(bbase + m)) * TLEN + t) * 3 + cc];
      st_bf(XA1, cc, m, x);
      st_bf(XA2, 496 + cc, m, x);
      st_bf(XA3, 496 + cc, m, x);
    }
    __syncthreads();

    // ---- LSTM 1: h1 -> XA1@32 (recurrence) and XA2@0 ---------------------
    lstm_stage<NKT1>(XA1, PW1, XA1, 32, XA2, 0, c1, lane, wave);

    // ---- window params: wp = exp(h1 @ w_win^T + b_win) -------------------
    if (wave < 2) {
      v8f acc = vz;
      for (int ktl = 0; ktl < 13; ++ktl) {
        v16bf a = *(const v16bf*)(XA2 + (((ktl * 32) + lane) << 4));
        v16bf b =
            *(const v16bf*)(PWW + ((size_t)((wave * 13 + ktl) * 32 + lane) << 4));
        acc = wmma_bf16(a, b, acc);
      }
      int col = wave * 16 + (lane & 15);
      if (col < 30) {
        int hi = lane >> 4;
#pragma unroll
        for (int e = 0; e < 8; ++e) {
          int m = e + 8 * hi;
          float v = __expf(acc[e]);
          if (col < 10)      sAlpha[m * 10 + col] = v;
          else if (col < 20) sBeta[m * 10 + col - 10] = v;
          else               sKappa[m * 10 + col - 20] += v;  // kappa += dkappa
        }
      }
    }
    __syncthreads();

    // ---- phi[m,u] = sum_k alpha * exp(-beta * (kappa-u)^2) ---------------
    for (int i = tid; i < 16 * ULEN; i += 256) {
      int m = i / ULEN, u = i - m * ULEN;
      float fu = (float)u, s = 0.f;
#pragma unroll
      for (int k = 0; k < KWIN; ++k) {
        float d = sKappa[m * 10 + k] - fu;
        s += sAlpha[m * 10 + k] * __expf(-sBeta[m * 10 + k] * d * d);
      }
      sPhi[i] = s;
    }
    __syncthreads();

    // ---- window[m,a] = phi[m,:] . charT[b,a,:] ---------------------------
    for (int i = tid; i < 16 * ALPHAN; i += 256) {
      int m = i / ALPHAN, a = i - m * ALPHAN;
      const float* p = csT + ((size_t)(bbase + m) * ALPHAN + a) * ULEN;
      const float* ph = sPhi + m * ULEN;
      float s = 0.f;
      for (int u = 0; u < ULEN; ++u) s += ph[u] * p[u];
      st_bf(XA2, 416 + a, m, s);
      st_bf(XA3, 416 + a, m, s);
    }
    __syncthreads();

    // ---- LSTM 2: h2 -> XA2@512 (recurrence) and XA3@0 --------------------
    lstm_stage<NKT23>(XA2, PW2, XA2, 512, XA3, 0, c2, lane, wave);

    // ---- LSTM 3: h3 -> XA3@512 (recurrence) ------------------------------
    lstm_stage<NKT23>(XA3, PW3, XA3, 512, (__bf16*)nullptr, 0, c3, lane, wave);

    // ---- MDN head: out = h3 @ w_mdn^T + b_mdn ----------------------------
    {
      v8f acc = vz;
      for (int ktl = 0; ktl < 13; ++ktl) {
        v16bf a = *(const v16bf*)(XA3 + ((((16 + ktl) * 32) + lane) << 4));
        v16bf b =
            *(const v16bf*)(PWM + ((size_t)((wave * 13 + ktl) * 32 + lane) << 4));
        acc = wmma_bf16(a, b, acc);
      }
      int col = wave * 16 + (lane & 15);
      if (col < MDNOUT) {
        int hi = lane >> 4;
#pragma unroll
        for (int e = 0; e < 8; ++e) {
          int m = e + 8 * hi;
          out[(((size_t)(bbase + m)) * TLEN + t) * MDNOUT + col] = acc[e];
        }
      }
    }
    // next-step x_t writes touch only disjoint K regions; no barrier needed here
  }
}

// ---------------------------------------------------------------------------
extern "C" void kernel_launch(void* const* d_in, const int* in_sizes, int n_in,
                              void* d_out, int out_size, void* d_ws,
                              size_t ws_size, hipStream_t stream) {
  const float* traj  = (const float*)d_in[0];
  const float* cs    = (const float*)d_in[1];
  const float* w_ih1 = (const float*)d_in[2];
  const float* w_hh1 = (const float*)d_in[3];
  const float* b1    = (const float*)d_in[4];
  const float* w_ih2 = (const float*)d_in[5];
  const float* w_hh2 = (const float*)d_in[6];
  const float* b2    = (const float*)d_in[7];
  const float* w_ih3 = (const float*)d_in[8];
  const float* w_hh3 = (const float*)d_in[9];
  const float* b3    = (const float*)d_in[10];
  const float* w_win = (const float*)d_in[11];
  const float* b_win = (const float*)d_in[12];
  const float* w_mdn = (const float*)d_in[13];
  const float* b_mdn = (const float*)d_in[14];
  float* out = (float*)d_out;

  char* ws = (char*)d_ws;
  size_t off = 0;
  void* PW1 = ws + off; off += (size_t)25 * 4 * NKT1 * 512 * 2;   // 1,433,600
  void* PW2 = ws + off; off += (size_t)25 * 4 * NKT23 * 512 * 2;  // 2,969,600
  void* PW3 = ws + off; off += (size_t)25 * 4 * NKT23 * 512 * 2;  // 2,969,600
  void* PWW = ws + off; off += (size_t)2 * 13 * 512 * 2;          //    26,624
  void* PWM = ws + off; off += (size_t)8 * 13 * 512 * 2;          //   106,496
  float* csT = (float*)(ws + off); off += (size_t)BTOT * ULEN * ALPHAN * 4;

  pack_lstm_k<<<1024, 256, 0, stream>>>(PW1, w_ih1, w_hh1, b1, NKT1, 1);
  pack_lstm_k<<<2048, 256, 0, stream>>>(PW2, w_ih2, w_hh2, b2, NKT23, 2);
  pack_lstm_k<<<2048, 256, 0, stream>>>(PW3, w_ih3, w_hh3, b3, NKT23, 2);
  pack_win_k<<<32, 256, 0, stream>>>(PWW, w_win, b_win);
  pack_mdn_k<<<64, 256, 0, stream>>>(PWM, w_mdn, b_mdn);
  tr_char_k<<<2048, 256, 0, stream>>>(csT, cs);

  const int smem_bytes = (NKT1 + 2 * NKT23) * 512 * 2 + (160 * 3 + 1600) * 4; // 82,048
  rnn_main<<<BTOT / 16, 256, smem_bytes, stream>>>(traj, csT, PW1, PW2, PW3,
                                                   PWW, PWM, out);
}
